// LocationSensitiveAttention_1425929142636
// MI455X (gfx1250) — matrix operand order repaired and checked by reading, not compiled, not run on hardware
//
#include <hip/hip_runtime.h>

#define B_ 32
#define S_ 4096
#define D_ 512
#define U_ 128
#define F_ 32
#define K_ 31

// 17 k-steps (16 for values@Wv, 1 for Toeplitz(prev)@CW), 8 n-tiles, 32 lanes, 8 dwords
#define WSB_WORDS (17 * 8 * 32 * 8)   // 34816 u32 = 136 KB

typedef __attribute__((ext_vector_type(16))) __bf16 v16bf;
typedef __attribute__((ext_vector_type(8)))  float  v8f;

union BF16x16 { unsigned int u[8]; v16bf v; };

__device__ __forceinline__ unsigned int f2bf_bits(float f) {
  unsigned int x = __float_as_uint(f);
  return (x + 0x7FFFu + ((x >> 16) & 1u)) >> 16;   // round-to-nearest-even
}

// --- pack B (Wv bf16, plus CW = conv_k@Wl for the 17th k-step) into CDNA5 B-lane layout ---
// word = ((ks*8 + t)*32 + L)*8 + j ; lane L: col n = 16t + (L&15), khalf = L>>4,
// dword j holds K = 32*ks + 8*khalf + 2j + 8*(j>=4) (low half) and +1 (high half).
__global__ void k_pack_B(const float* __restrict__ Wv,
                         const float* __restrict__ conv_k,
                         const float* __restrict__ Wl,
                         unsigned int* __restrict__ wsB) {
  int w = blockIdx.x * 256 + threadIdx.x;
  if (w >= WSB_WORDS) return;
  int j  = w & 7;
  int L  = (w >> 3) & 31;
  int t  = (w >> 8) & 7;
  int ks = w >> 11;
  int n     = 16 * t + (L & 15);
  int khalf = L >> 4;
  int kloc  = 8 * khalf + 2 * j + (j >= 4 ? 8 : 0);
  float v0 = 0.f, v1 = 0.f;
  if (ks < 16) {
    int k0 = 32 * ks + kloc;
    v0 = Wv[(size_t)k0 * U_ + n];
    v1 = Wv[(size_t)(k0 + 1) * U_ + n];
  } else {
    // CW[k][n] = sum_f conv_k[k,0,f] * Wl[f,n] ; row 31 padded with zero
    if (kloc < K_) {
      float a = 0.f;
      for (int f = 0; f < F_; ++f) a += conv_k[kloc * F_ + f] * Wl[f * U_ + n];
      v0 = a;
    }
    if (kloc + 1 < K_) {
      float a = 0.f;
      for (int f = 0; f < F_; ++f) a += conv_k[(kloc + 1) * F_ + f] * Wl[f * U_ + n];
      v1 = a;
    }
  }
  wsB[w] = f2bf_bits(v0) | (f2bf_bits(v1) << 16);
}

// --- base[b,u] = query@Wq + bq + bv + bl (tiny: 4096 dots of 512) ---
__global__ void k_base(const float* __restrict__ query, const float* __restrict__ Wq,
                       const float* __restrict__ bq, const float* __restrict__ bv,
                       const float* __restrict__ bl, float* __restrict__ base) {
  int idx = blockIdx.x * 256 + threadIdx.x;      // 4096 total
  int b = idx >> 7, u = idx & 127;
  float a = bq[u] + bv[u] + bl[u];
  const float* qr = query + b * D_;
  for (int i = 0; i < D_; ++i) a += qr[i] * Wq[i * U_ + u];
  base[idx] = a;
}

// --- main score kernel: per wave a 16x128 strip, K=544 via WMMA, fused tanh+Vw reduce ---
__global__ __launch_bounds__(256)
void k_score(const float* __restrict__ values,
             const float* __restrict__ prev,
             const unsigned char* __restrict__ mask,
             const unsigned int* __restrict__ wsB,
             const float* __restrict__ base,
             const float* __restrict__ Vw,
             const float* __restrict__ Vb,
             float* __restrict__ score) {
  __shared__ float prevw[160];                   // prev[s0-15 .. s0+143], zero-padded
  const int b   = blockIdx.y;
  const int s0  = blockIdx.x * 128;
  const int tid = threadIdx.x;
  const int w     = tid >> 5;                    // wave 0..7 -> rows 16w..16w+15
  const int L     = tid & 31;
  const int nlane = L & 15;
  const int khalf = L >> 4;

  for (int i = tid; i < 159; i += 256) {
    int s = s0 - 15 + i;
    prevw[i] = (s >= 0 && s < S_) ? prev[b * S_ + s] : 0.f;
  }
  if (tid == 0) prevw[159] = 0.f;
  __syncthreads();

  v8f c[8];
  #pragma unroll
  for (int t = 0; t < 8; ++t)
    #pragma unroll
    for (int r = 0; r < 8; ++r) c[t][r] = 0.f;

  const int row = 16 * w + nlane;                // A-matrix row within workgroup tile
  const float* rp = values + (size_t)(b * S_ + s0 + row) * D_;

  for (int ks = 0; ks < 17; ++ks) {
    float fA[16];
    if (ks < 16) {
      const float4* p0 = (const float4*)(rp + ks * 32 + 8 * khalf);
      const float4* p1 = (const float4*)(rp + ks * 32 + 16 + 8 * khalf);
      float4 f0 = p0[0], f1 = p0[1], f2 = p1[0], f3 = p1[1];
      fA[0] = f0.x; fA[1] = f0.y; fA[2]  = f0.z; fA[3]  = f0.w;
      fA[4] = f1.x; fA[5] = f1.y; fA[6]  = f1.z; fA[7]  = f1.w;
      fA[8] = f2.x; fA[9] = f2.y; fA[10] = f2.z; fA[11] = f2.w;
      fA[12] = f3.x; fA[13] = f3.y; fA[14] = f3.z; fA[15] = f3.w;
    } else {                                     // Toeplitz prev window -> location term
      #pragma unroll
      for (int e = 0; e < 8; ++e) {
        fA[e]     = prevw[row + 8 * khalf + e];
        fA[8 + e] = prevw[row + 16 + 8 * khalf + e];
      }
    }
    // split A into hi/lo bf16 (error-compensated, ~fp32-accurate with fp32 accum)
    BF16x16 ah, al;
    #pragma unroll
    for (int j = 0; j < 8; ++j) {
      float f0 = fA[2 * j], f1 = fA[2 * j + 1];
      unsigned int h0 = f2bf_bits(f0), h1 = f2bf_bits(f1);
      float r0 = f0 - __uint_as_float(h0 << 16);
      float r1 = f1 - __uint_as_float(h1 << 16);
      ah.u[j] = h0 | (h1 << 16);
      al.u[j] = f2bf_bits(r0) | (f2bf_bits(r1) << 16);
    }
    #pragma unroll
    for (int t = 0; t < 8; ++t) {
      BF16x16 bb;
      const uint4* bp = (const uint4*)(wsB + (((ks * 8 + t) * 32 + L) << 3));
      uint4 q0 = bp[0], q1 = bp[1];
      bb.u[0] = q0.x; bb.u[1] = q0.y; bb.u[2] = q0.z; bb.u[3] = q0.w;
      bb.u[4] = q1.x; bb.u[5] = q1.y; bb.u[6] = q1.z; bb.u[7] = q1.w;
      c[t] = __builtin_amdgcn_wmma_f32_16x16x32_bf16(false, ah.v, false, bb.v,
                                                     (short)0, c[t], false, false);
      c[t] = __builtin_amdgcn_wmma_f32_16x16x32_bf16(false, al.v, false, bb.v,
                                                     (short)0, c[t], false, false);
    }
  }

  // finalize: pre -> tanh -> dot Vw -> reduce across 16-lane halves -> masked score
  float Vb0 = Vb[0];
  float pr[8];
  #pragma unroll
  for (int r = 0; r < 8; ++r) pr[r] = 0.f;
  #pragma unroll
  for (int t = 0; t < 8; ++t) {
    int n = 16 * t + nlane;
    float bs = base[b * U_ + n];
    float vw = Vw[n];
    #pragma unroll
    for (int r = 0; r < 8; ++r) pr[r] += tanhf(c[t][r] + bs) * vw;
  }
  #pragma unroll
  for (int r = 0; r < 8; ++r) {
    float v = pr[r];
    v += __shfl_xor(v, 1);
    v += __shfl_xor(v, 2);
    v += __shfl_xor(v, 4);
    v += __shfl_xor(v, 8);                       // reduce within each 16-lane half
    if (nlane == 0) {                            // lanes 0 (rows r) and 16 (rows r+8)
      int srow = s0 + 16 * w + 8 * khalf + r;
      float sc = v + Vb0;
      if (!mask[b * S_ + srow]) sc -= 1e9f;
      score[b * S_ + srow] = sc;
    }
  }
}

// --- in-place softmax over S per batch row ---
__global__ void k_softmax(float* __restrict__ attn) {
  __shared__ float red[256];
  int b = blockIdx.x, tid = threadIdx.x;
  float* a = attn + b * S_;
  float mx = -3.4e38f;
  for (int s = tid; s < S_; s += 256) mx = fmaxf(mx, a[s]);
  red[tid] = mx; __syncthreads();
  for (int st = 128; st > 0; st >>= 1) {
    if (tid < st) red[tid] = fmaxf(red[tid], red[tid + st]);
    __syncthreads();
  }
  mx = red[0]; __syncthreads();
  float sum = 0.f;
  for (int s = tid; s < S_; s += 256) { float e = __expf(a[s] - mx); a[s] = e; sum += e; }
  red[tid] = sum; __syncthreads();
  for (int st = 128; st > 0; st >>= 1) {
    if (tid < st) red[tid] += red[tid + st];
    __syncthreads();
  }
  float inv = 1.f / red[0];
  for (int s = tid; s < S_; s += 256) a[s] *= inv;
}

// --- context = attn @ values : split over S for occupancy, then reduce ---
__global__ void k_ctx_partial(const float* __restrict__ values,
                              const float* __restrict__ attn,
                              float* __restrict__ part) {
  int b = blockIdx.y, chunk = blockIdx.x, d = threadIdx.x;   // 512 threads = D
  int sbase = chunk * 512;
  const float* vp = values + ((size_t)b * S_ + sbase) * D_ + d;
  const float* ap = attn + b * S_ + sbase;
  float acc = 0.f;
  for (int i = 0; i < 512; ++i) acc += ap[i] * vp[(size_t)i * D_];
  part[(b * 8 + chunk) * 512 + d] = acc;
}

__global__ void k_ctx_reduce(const float* __restrict__ part, float* __restrict__ ctx) {
  int b = blockIdx.x, d = threadIdx.x;
  float acc = 0.f;
  for (int ch = 0; ch < 8; ++ch) acc += part[(b * 8 + ch) * 512 + d];
  ctx[b * 512 + d] = acc;
}

extern "C" void kernel_launch(void* const* d_in, const int* in_sizes, int n_in,
                              void* d_out, int out_size, void* d_ws, size_t ws_size,
                              hipStream_t stream) {
  const float* query  = (const float*)d_in[0];
  const float* values = (const float*)d_in[1];
  const float* prev   = (const float*)d_in[2];
  const unsigned char* mask = (const unsigned char*)d_in[3];
  const float* Wq = (const float*)d_in[4];
  const float* bq = (const float*)d_in[5];
  const float* Wv = (const float*)d_in[6];
  const float* bv = (const float*)d_in[7];
  const float* Wl = (const float*)d_in[8];
  const float* bl = (const float*)d_in[9];
  const float* Vw = (const float*)d_in[10];
  const float* Vb = (const float*)d_in[11];
  const float* conv_k = (const float*)d_in[12];

  float* ctx  = (float*)d_out;                 // [B, D]
  float* attn = ctx + B_ * D_;                 // [B, S] (scores -> softmax in place)

  unsigned int* wsB = (unsigned int*)d_ws;     // 136 KB packed B
  float* basep = (float*)d_ws + WSB_WORDS;     // 16 KB base[b,u]
  float* part  = basep + B_ * U_;              // 512 KB context partials

  k_pack_B<<<WSB_WORDS / 256, 256, 0, stream>>>(Wv, conv_k, Wl, wsB);
  k_base<<<16, 256, 0, stream>>>(query, Wq, bq, bv, bl, basep);
  k_score<<<dim3(32, 32), 256, 0, stream>>>(values, prev, mask, wsB, basep, Vw, Vb, attn);
  k_softmax<<<32, 256, 0, stream>>>(attn);
  k_ctx_partial<<<dim3(8, 32), 512, 0, stream>>>(values, attn, part);
  k_ctx_reduce<<<32, 512, 0, stream>>>(part, ctx);
}
